// Block_49529562858355
// MI455X (gfx1250) — matrix operand order repaired and checked by reading, not compiled, not run on hardware
//
#include <hip/hip_runtime.h>

// ---------------------------------------------------------------------------
// DiT adaLN block forward for MI455X (gfx1250).
// bf16 WMMA + f32 accumulation; weight tiles streamed to LDS via the
// Tensor Data Mover (TENSORcnt double buffering); register-pipelined A loads.
// ---------------------------------------------------------------------------

constexpr int Bc   = 2;
constexpr int Nc   = 2048;
constexpr int Dm   = 1024;
constexpr int Hh   = 16;
constexpr int DHc  = 64;
constexpr int HIDc = 4096;
constexpr int SIXD   = 6 * Dm;   // 6144
constexpr int THREED = 3 * Dm;   // 3072
constexpr int BN     = Bc * Nc;  // 4096

typedef __bf16 v16bf __attribute__((ext_vector_type(16)));
typedef float  v8f   __attribute__((ext_vector_type(8)));
typedef unsigned int u32x4 __attribute__((ext_vector_type(4)));
typedef int          i32x4 __attribute__((ext_vector_type(4)));
typedef int          i32x8 __attribute__((ext_vector_type(8)));

struct U128 { unsigned int x, y, z, w; };
union VPack { v16bf v; U128 u[2]; __bf16 e[16]; };

#define DEV __device__ __forceinline__

DEV __bf16 f2bf(float f) {
  unsigned u = __builtin_bit_cast(unsigned, f);
  unsigned r = u + 0x7FFFu + ((u >> 16) & 1u);   // round-to-nearest-even
  unsigned short h = (unsigned short)(r >> 16);
  return __builtin_bit_cast(__bf16, h);
}

#define WMMA_BF16(a, b, c) \
  __builtin_amdgcn_wmma_f32_16x16x32_bf16(false, (a), false, (b), (short)0, (c), false, false)

// A-operand (16x32 bf16): lane holds row m=lane%16; elements [0..7] = K(k0+half*8..+7),
// elements [8..15] = K(k0+16+half*8..+7). Two contiguous 16B chunks.
DEV v16bf load_a16(const __bf16* __restrict__ row, int k0, int half) {
  VPack t;
  t.u[0] = *reinterpret_cast<const U128*>(row + k0 + half * 8);
  t.u[1] = *reinterpret_cast<const U128*>(row + k0 + 16 + half * 8);
  return t.v;
}

// B-operand (32x16 bf16): lane holds column n=lane%16 (a row of the transposed
// weight); elements [0..7] = K(k0+half*8..+7), [8..15] = K(k0+16+half*8..+7).
DEV v16bf load_b16(const __bf16* __restrict__ row, int k0, int half) {
  VPack t;
  const __bf16* p = row + k0 + half * 8;
  t.u[0] = *reinterpret_cast<const U128*>(p);
  t.u[1] = *reinterpret_cast<const U128*>(p + 16);
  return t.v;
}

// ---------------------- Tensor Data Mover (2D tile -> LDS) -----------------
// D# per CDNA5 ISA: group0 = {count/flags, lds_addr, global_addr, type=2},
// group1 = {data_size=2B, tensor_dim0=K, tensor_dim1=rows, tile 64(k) x 64(rows),
//           tensor_dim0_stride=K}. Groups 2/3 zero (2D tensor).
// This toolchain's builtin is the 6-arg form:
//   (u32x4 g0, i32x8 g1, i32x4 g2, i32x4 g3, i32x8 g4, i32 cpol)
DEV void tdm_load_tile(unsigned lds_off, const __bf16* gsrc, int K, int tensorRows) {
  unsigned long long ga = (unsigned long long)(uintptr_t)gsrc;
  u32x4 g0 = { 0x1u,                                   // count=1, user mode
               lds_off,                                 // lds_addr (bytes)
               (unsigned)(ga & 0xFFFFFFFFu),            // global_addr[31:0]
               (unsigned)((ga >> 32) & 0x1FFFFFFu) | (2u << 30) };  // addr msbs | type=2
  unsigned td0 = (unsigned)K;           // tensor_dim0 (elements per row)
  unsigned td1 = (unsigned)tensorRows;  // tensor_dim1 (rows)
  i32x8 g1 = { (int)0x10000u,                                   // data_size=1 (2 bytes)
               (int)((td0 & 0xFFFFu) << 16),                    // tensor_dim0 lo16 @bit48
               (int)((td0 >> 16) | ((td1 & 0xFFFFu) << 16)),    // dim0 hi16 | dim1 lo16
               (int)((td1 >> 16) | (64u << 16)),                // dim1 hi16 | tile_dim0=64
               (int)64,                                         // tile_dim1=64, tile_dim2=0
               (int)td0,                                        // tensor_dim0_stride lo32
               0, 0 };
  i32x4 z4 = { 0, 0, 0, 0 };
  i32x8 z8 = { 0, 0, 0, 0, 0, 0, 0, 0 };
  __builtin_amdgcn_tensor_load_to_lds(g0, g1, z4, z4, z8, 0);
}

// ---------------------- LDS-staged GEMM core -------------------------------
// Workgroup: 4 waves -> 64(M) x 64(N) tile. B tile (64 rows x 64 k bf16, 8KB)
// double-buffered in LDS via TDM; per-wave A operands register-pipelined.
DEV void gemm_lds_core(const __bf16* __restrict__ A, const __bf16* __restrict__ BT,
                       int K, int tensorRows, int mRowWave, int nBase,
                       __bf16* bsm, int lane16, int half, int wave, v8f acc[4]) {
  const __bf16* arow = A + (size_t)(mRowWave + lane16) * K;
  const int nchunks = K / 64;
  if (wave == 0) tdm_load_tile((unsigned)(uintptr_t)bsm, BT + (size_t)nBase * K, K, tensorRows);
  v16bf a0 = load_a16(arow, 0, half);
  v16bf a1 = load_a16(arow, 32, half);
  for (int c = 0; c < nchunks; ++c) {
    const __bf16* buf = bsm + (c & 1) * (64 * 64);
    if (wave == 0) {
      if (c + 1 < nchunks) {
        tdm_load_tile((unsigned)(uintptr_t)(bsm + ((c + 1) & 1) * (64 * 64)),
                      BT + (size_t)nBase * K + (c + 1) * 64, K, tensorRows);
        __builtin_amdgcn_s_wait_tensorcnt((short)1);   // chunk c landed, c+1 in flight
      } else {
        __builtin_amdgcn_s_wait_tensorcnt((short)0);
      }
    }
    v16bf na0 = a0, na1 = a1;
    if (c + 1 < nchunks) {                             // prefetch next A chunk
      na0 = load_a16(arow, (c + 1) * 64, half);
      na1 = load_a16(arow, (c + 1) * 64 + 32, half);
    }
    __syncthreads();                                   // B tile ready for all waves
#pragma unroll
    for (int nt = 0; nt < 4; ++nt) {
      const __bf16* brow = buf + (nt * 16 + lane16) * 64;
      acc[nt] = WMMA_BF16(a0, load_b16(brow, 0, half), acc[nt]);
      acc[nt] = WMMA_BF16(a1, load_b16(brow, 32, half), acc[nt]);
    }
    __syncthreads();                                   // done reading before overwrite
    a0 = na0; a1 = na1;
  }
}

// ------------------------------- prep kernels ------------------------------

__global__ void silu_kernel(const float* __restrict__ cond, float* __restrict__ out) {
  int i = blockIdx.x * 256 + threadIdx.x;
  if (i < Bc * Dm) { float v = cond[i]; out[i] = v / (1.0f + __expf(-v)); }
}

__global__ void ada_kernel(const float* __restrict__ siluc, const float* __restrict__ w_ada,
                           const float* __restrict__ b_ada, float* __restrict__ ada) {
  int i = blockIdx.x * 256 + threadIdx.x;
  if (i >= Bc * SIXD) return;
  int b = i / SIXD, c = i % SIXD;
  float acc = b_ada[c];
  const float* sc = siluc + b * Dm;
  for (int k = 0; k < Dm; ++k) acc += sc[k] * w_ada[(size_t)k * SIXD + c];
  ada[i] = acc;
}

// transpose + cast: in[R][C] f32 -> out[C][R] bf16
__global__ void castT_kernel(const float* __restrict__ in, __bf16* __restrict__ out,
                             int rows, int cols) {
  size_t i = (size_t)blockIdx.x * 256 + threadIdx.x;
  if (i >= (size_t)rows * cols) return;
  int r = (int)(i / cols), c = (int)(i % cols);
  out[(size_t)c * rows + r] = f2bf(in[i]);
}

DEV float block_sum(float v, float* red) {
  int t = threadIdx.x;
  red[t] = v; __syncthreads();
  for (int s = 128; s > 0; s >>= 1) { if (t < s) red[t] += red[t + s]; __syncthreads(); }
  float r = red[0]; __syncthreads();
  return r;
}

// h1 = LN(x,1e-6)*(1+sc_msa)+sh_msa ; hA = LN(h1,1e-5)*ln_g+ln_b  (bf16)
__global__ void ln1_kernel(const float* __restrict__ x, const float* __restrict__ ada,
                           const float* __restrict__ ln_g, const float* __restrict__ ln_b,
                           __bf16* __restrict__ hA) {
  __shared__ float red[256];
  const int row = blockIdx.x;
  const int b = row >> 11;
  const int t = threadIdx.x;
  const float* xr = x + (size_t)row * Dm;
  const float* ab = ada + (size_t)b * SIXD;
  float v[4];
#pragma unroll
  for (int i = 0; i < 4; ++i) v[i] = xr[t + 256 * i];
  float mean = block_sum(v[0] + v[1] + v[2] + v[3], red) * (1.0f / Dm);
  float q = 0.f;
#pragma unroll
  for (int i = 0; i < 4; ++i) { float d = v[i] - mean; q += d * d; }
  float rstd = rsqrtf(block_sum(q, red) * (1.0f / Dm) + 1e-6f);
  float h[4];
#pragma unroll
  for (int i = 0; i < 4; ++i) {
    int c = t + 256 * i;
    h[i] = (v[i] - mean) * rstd * (1.0f + ab[Dm + c]) + ab[c];
  }
  float mean2 = block_sum(h[0] + h[1] + h[2] + h[3], red) * (1.0f / Dm);
  float q2 = 0.f;
#pragma unroll
  for (int i = 0; i < 4; ++i) { float d = h[i] - mean2; q2 += d * d; }
  float rstd2 = rsqrtf(block_sum(q2, red) * (1.0f / Dm) + 1e-5f);
#pragma unroll
  for (int i = 0; i < 4; ++i) {
    int c = t + 256 * i;
    hA[(size_t)row * Dm + c] = f2bf((h[i] - mean2) * rstd2 * ln_g[c] + ln_b[c]);
  }
}

// x1 = x + g_msa*attn ; h2 = LN(x1,1e-6)*(1+sc_mlp)+sh_mlp (bf16)
__global__ void fuse2_kernel(const float* __restrict__ x, const float* __restrict__ attn,
                             const float* __restrict__ ada, float* __restrict__ x1,
                             __bf16* __restrict__ h2) {
  __shared__ float red[256];
  const int row = blockIdx.x;
  const int b = row >> 11;
  const int t = threadIdx.x;
  const float* ab = ada + (size_t)b * SIXD;
  float v[4];
#pragma unroll
  for (int i = 0; i < 4; ++i) {
    int c = t + 256 * i;
    size_t idx = (size_t)row * Dm + c;
    float val = x[idx] + ab[2 * Dm + c] * attn[idx];
    x1[idx] = val;
    v[i] = val;
  }
  float mean = block_sum(v[0] + v[1] + v[2] + v[3], red) * (1.0f / Dm);
  float q = 0.f;
#pragma unroll
  for (int i = 0; i < 4; ++i) { float d = v[i] - mean; q += d * d; }
  float rstd = rsqrtf(block_sum(q, red) * (1.0f / Dm) + 1e-6f);
#pragma unroll
  for (int i = 0; i < 4; ++i) {
    int c = t + 256 * i;
    h2[(size_t)row * Dm + c] = f2bf((v[i] - mean) * rstd * (1.0f + ab[4 * Dm + c]) + ab[3 * Dm + c]);
  }
}

// ------------------------------- GEMM kernels ------------------------------

// QKV: hA[BN,1024] x wqkvT -> scatter Q (scaled 1/8), K row-major, V transposed.
__global__ void gemm_qkv_kernel(const __bf16* __restrict__ A, const __bf16* __restrict__ BT,
                                __bf16* __restrict__ Qb, __bf16* __restrict__ Kb,
                                __bf16* __restrict__ Vt) {
  __shared__ __bf16 bsm[2 * 64 * 64];
  const int tid = threadIdx.x;
  const int lane16 = tid & 15, half = (tid >> 4) & 1, wave = tid >> 5;
  const int tilesN = THREED / 64;                    // 48
  const int bm = blockIdx.x / tilesN, bn = blockIdx.x % tilesN;
  const int mRow = bm * 64 + wave * 16, nBase = bn * 64;
  v8f acc[4] = {};
  gemm_lds_core(A, BT, Dm, THREED, mRow, nBase, bsm, lane16, half, wave, acc);
#pragma unroll
  for (int nt = 0; nt < 4; ++nt) {
    int c = nBase + nt * 16 + lane16;
    int which = c >> 10;                             // 0=q,1=k,2=v (wave-uniform)
    int cc = c & 1023, hh = cc >> 6, d = cc & 63;
#pragma unroll
    for (int i = 0; i < 8; ++i) {
      int m = mRow + i + 8 * half;
      int b = m >> 11, n = m & 2047;
      size_t bh = (size_t)b * Hh + hh;
      float v = acc[nt][i];
      if (which == 0)      Qb[(bh * Nc + n) * DHc + d] = f2bf(v * 0.125f);
      else if (which == 1) Kb[(bh * Nc + n) * DHc + d] = f2bf(v);
      else                 Vt[(bh * DHc + d) * Nc + n] = f2bf(v);
    }
  }
}

// MLP1: h2[BN,1024] x w1T -> +b1, tanh-GELU -> bf16 hm[BN,4096]
__global__ void gemm_mlp1_kernel(const __bf16* __restrict__ A, const __bf16* __restrict__ BT,
                                 const float* __restrict__ b1, __bf16* __restrict__ hm) {
  __shared__ __bf16 bsm[2 * 64 * 64];
  const int tid = threadIdx.x;
  const int lane16 = tid & 15, half = (tid >> 4) & 1, wave = tid >> 5;
  const int tilesN = HIDc / 64;                      // 64
  const int bm = blockIdx.x / tilesN, bn = blockIdx.x % tilesN;
  const int mRow = bm * 64 + wave * 16, nBase = bn * 64;
  v8f acc[4] = {};
  gemm_lds_core(A, BT, Dm, HIDc, mRow, nBase, bsm, lane16, half, wave, acc);
#pragma unroll
  for (int nt = 0; nt < 4; ++nt) {
    int c = nBase + nt * 16 + lane16;
    float bias = b1[c];
#pragma unroll
    for (int i = 0; i < 8; ++i) {
      int m = mRow + i + 8 * half;
      float v = acc[nt][i] + bias;
      float g = 0.5f * v * (1.0f + tanhf(0.7978845608f * (v + 0.044715f * v * v * v)));
      hm[(size_t)m * HIDc + c] = f2bf(g);
    }
  }
}

// MLP2: hm[BN,4096] x w2T -> +b2 ; out = x1 + g_mlp * y
__global__ void gemm_mlp2_kernel(const __bf16* __restrict__ A, const __bf16* __restrict__ BT,
                                 const float* __restrict__ b2, const float* __restrict__ x1,
                                 const float* __restrict__ ada, float* __restrict__ out) {
  __shared__ __bf16 bsm[2 * 64 * 64];
  const int tid = threadIdx.x;
  const int lane16 = tid & 15, half = (tid >> 4) & 1, wave = tid >> 5;
  const int tilesN = Dm / 64;                        // 16
  const int bm = blockIdx.x / tilesN, bn = blockIdx.x % tilesN;
  const int mRow = bm * 64 + wave * 16, nBase = bn * 64;
  v8f acc[4] = {};
  gemm_lds_core(A, BT, HIDc, Dm, mRow, nBase, bsm, lane16, half, wave, acc);
#pragma unroll
  for (int nt = 0; nt < 4; ++nt) {
    int c = nBase + nt * 16 + lane16;
    float bias = b2[c];
#pragma unroll
    for (int i = 0; i < 8; ++i) {
      int m = mRow + i + 8 * half;
      int b = m >> 11;
      float y = acc[nt][i] + bias;
      size_t idx = (size_t)m * Dm + c;
      out[idx] = x1[idx] + ada[(size_t)b * SIXD + 5 * Dm + c] * y;
    }
  }
}

// ------------------------------- attention ---------------------------------

constexpr int SROW = Nc + 4;                         // padded LDS row (bank spread)
constexpr unsigned SMEM_ATTN = (16 * SROW + 16) * sizeof(float);

__global__ void attn_kernel(const __bf16* __restrict__ Qb, const __bf16* __restrict__ Kb,
                            const __bf16* __restrict__ Vt, const int* __restrict__ mask,
                            float* __restrict__ attn_out) {
  extern __shared__ float smem[];
  float* S  = smem;
  float* RS = smem + 16 * SROW;
  const int lane = threadIdx.x & 31, lane16 = lane & 15, half = lane >> 4;
  const int blk = blockIdx.x;                        // b*H*(N/16)
  const int mt = blk & 127;                          // N/16 = 128
  const int bh = blk >> 7;                           // b*H + h
  const int b = bh >> 4, hh = bh & 15;

  // Q tile (A operand), loaded once; q pre-scaled by DH^-0.5 at QKV store.
  const __bf16* Qrow = Qb + ((size_t)bh * Nc + mt * 16 + lane16) * DHc;
  v16bf aq0 = load_a16(Qrow, 0, half);
  v16bf aq1 = load_a16(Qrow, 32, half);

  // S = Q K^T : 128 key tiles, register-pipelined K operands.
  const __bf16* Kbase = Kb + ((size_t)bh * Nc + lane16) * DHc;
  v16bf bk0 = load_b16(Kbase, 0, half);
  v16bf bk1 = load_b16(Kbase, 32, half);
  for (int jt = 0; jt < Nc / 16; ++jt) {
    v16bf nb0 = bk0, nb1 = bk1;
    if (jt + 1 < Nc / 16) {
      const __bf16* Krow = Kbase + (size_t)(jt + 1) * 16 * DHc;
      nb0 = load_b16(Krow, 0, half);
      nb1 = load_b16(Krow, 32, half);
    }
    v8f c = {};
    c = WMMA_BF16(aq0, bk0, c);
    c = WMMA_BF16(aq1, bk1, c);
#pragma unroll
    for (int i = 0; i < 8; ++i)
      S[(i + 8 * half) * SROW + jt * 16 + lane16] = c[i];
    bk0 = nb0; bk1 = nb1;
  }
  __syncthreads();

  // Masked softmax: lane handles row lane16, columns [half*1024, +1024).
  {
    const int r = lane16, c0 = half * (Nc / 2);
    const int* mrow = mask + b * Nc;
    float* Srow = S + r * SROW;
    float mx = -3.4e38f;
    for (int j = c0; j < c0 + Nc / 2; ++j) {
      float v = (mrow[j] != 0) ? -3.4e38f : Srow[j];
      Srow[j] = v;
      mx = fmaxf(mx, v);
    }
    mx = fmaxf(mx, __shfl_xor(mx, 16));
    float sum = 0.f;
    for (int j = c0; j < c0 + Nc / 2; ++j) {
      float e = __expf(Srow[j] - mx);
      Srow[j] = e;
      sum += e;
    }
    sum += __shfl_xor(sum, 16);
    if (half == 0) RS[r] = sum;
  }
  __syncthreads();

  // O = P V : P (bf16 from LDS) as A operand, Vt rows as B operand.
  v8f co[4] = {};
  for (int kt = 0; kt < Nc / 32; ++kt) {
    const float* Sr = S + lane16 * SROW + kt * 32;
    VPack t;
#pragma unroll
    for (int e = 0; e < 8; ++e) {
      t.e[e]     = f2bf(Sr[half * 8 + e]);
      t.e[e + 8] = f2bf(Sr[16 + half * 8 + e]);
    }
    v16bf ap = t.v;
#pragma unroll
    for (int dt = 0; dt < 4; ++dt) {
      const __bf16* Vrow = Vt + ((size_t)bh * DHc + dt * 16 + lane16) * Nc;
      co[dt] = WMMA_BF16(ap, load_b16(Vrow, kt * 32, half), co[dt]);
    }
  }
#pragma unroll
  for (int dt = 0; dt < 4; ++dt) {
#pragma unroll
    for (int i = 0; i < 8; ++i) {
      int m = i + 8 * half;
      int tok = mt * 16 + m;
      float val = co[dt][i] / RS[m];
      attn_out[((size_t)b * Nc + tok) * Dm + hh * DHc + dt * 16 + lane16] = val;
    }
  }
}

// ------------------------------- launcher ----------------------------------

extern "C" void kernel_launch(void* const* d_in, const int* in_sizes, int n_in,
                              void* d_out, int out_size, void* d_ws, size_t ws_size,
                              hipStream_t stream) {
  const float* x     = (const float*)d_in[0];
  const float* cond  = (const float*)d_in[1];
  const int*   mask  = (const int*)d_in[2];
  const float* w_qkv = (const float*)d_in[3];
  const float* ln_g  = (const float*)d_in[4];
  const float* ln_b  = (const float*)d_in[5];
  const float* w1    = (const float*)d_in[6];
  const float* b1    = (const float*)d_in[7];
  const float* w2    = (const float*)d_in[8];
  const float* b2    = (const float*)d_in[9];
  const float* w_ada = (const float*)d_in[10];
  const float* b_ada = (const float*)d_in[11];
  float* out = (float*)d_out;
  char* ws = (char*)d_ws;

  auto al = [](size_t v) { return (v + 255) & ~(size_t)255; };
  size_t off = 0;
  float*  silu_c = (float*)(ws + off);  off = al(off + sizeof(float) * Bc * Dm);
  float*  ada    = (float*)(ws + off);  off = al(off + sizeof(float) * Bc * SIXD);
  __bf16* wqkvT  = (__bf16*)(ws + off); off = al(off + sizeof(__bf16) * (size_t)THREED * Dm);
  __bf16* w1T    = (__bf16*)(ws + off); off = al(off + sizeof(__bf16) * (size_t)HIDc * Dm);
  __bf16* w2T    = (__bf16*)(ws + off); off = al(off + sizeof(__bf16) * (size_t)Dm * HIDc);
  __bf16* hA     = (__bf16*)(ws + off); off = al(off + sizeof(__bf16) * (size_t)BN * Dm);
  __bf16* Qb     = (__bf16*)(ws + off); off = al(off + sizeof(__bf16) * (size_t)Bc * Hh * Nc * DHc);
  __bf16* Kb     = (__bf16*)(ws + off); off = al(off + sizeof(__bf16) * (size_t)Bc * Hh * Nc * DHc);
  __bf16* Vt     = (__bf16*)(ws + off); off = al(off + sizeof(__bf16) * (size_t)Bc * Hh * Nc * DHc);
  float*  attn   = (float*)(ws + off);  off = al(off + sizeof(float) * (size_t)BN * Dm);
  float*  x1     = (float*)(ws + off);  off = al(off + sizeof(float) * (size_t)BN * Dm);
  __bf16* h2     = (__bf16*)(ws + off); off = al(off + sizeof(__bf16) * (size_t)BN * Dm);
  __bf16* hm     = (__bf16*)(ws + off); off = al(off + sizeof(__bf16) * (size_t)BN * HIDc);
  (void)off; (void)in_sizes; (void)n_in; (void)out_size; (void)ws_size;

  (void)hipFuncSetAttribute(reinterpret_cast<const void*>(attn_kernel),
                            hipFuncAttributeMaxDynamicSharedMemorySize, (int)SMEM_ATTN);

  silu_kernel<<<(Bc * Dm + 255) / 256, 256, 0, stream>>>(cond, silu_c);
  ada_kernel<<<(Bc * SIXD + 255) / 256, 256, 0, stream>>>(silu_c, w_ada, b_ada, ada);

  castT_kernel<<<((size_t)Dm * THREED) / 256, 256, 0, stream>>>(w_qkv, wqkvT, Dm, THREED);
  castT_kernel<<<((size_t)Dm * HIDc) / 256, 256, 0, stream>>>(w1, w1T, Dm, HIDc);
  castT_kernel<<<((size_t)HIDc * Dm) / 256, 256, 0, stream>>>(w2, w2T, HIDc, Dm);

  ln1_kernel<<<BN, 256, 0, stream>>>(x, ada, ln_g, ln_b, hA);

  // QKV: (BN/64)*(3D/64) = 64*48 blocks, 4 waves each
  gemm_qkv_kernel<<<64 * 48, 128, 0, stream>>>(hA, wqkvT, Qb, Kb, Vt);

  attn_kernel<<<Bc * Hh * (Nc / 16), 32, SMEM_ATTN, stream>>>(Qb, Kb, Vt, mask, attn);

  fuse2_kernel<<<BN, 256, 0, stream>>>(x, attn, ada, x1, h2);

  gemm_mlp1_kernel<<<64 * 64, 128, 0, stream>>>(h2, w1T, b1, hm);
  gemm_mlp2_kernel<<<64 * 16, 128, 0, stream>>>(hm, w2T, b2, x1, ada, out);
}
